// SHGNN_nc_mb_81887846466108
// MI455X (gfx1250) — matrix operand order repaired and checked by reading, not compiled, not run on hardware
//
#include <hip/hip_runtime.h>

// SHGNN target-node fused kernel for gfx1250 (MI455X).
// Only the T=50000 target rows are computed (reference computes all ~1M and
// discards 95%): per-row type-selected feature GEMM via masked f16 WMMA,
// degree path as a 2-row gather-sum, and the 128x64 output GEMM via f16 WMMA.

namespace {
constexpr int kPerType = 333333;
constexpr int kF   = 64;
constexpr int kOut = 64;
constexpr int kD1  = 100;
constexpr int kD2  = 100;
constexpr int kT   = 50000;
constexpr int kWavesPerBlock = 4;
}

typedef __attribute__((ext_vector_type(16))) _Float16 v16h;
typedef __attribute__((ext_vector_type(8)))  _Float16 v8h;
typedef __attribute__((ext_vector_type(8)))  float    v8f;
typedef __attribute__((ext_vector_type(4)))  float    v4f;

__device__ __forceinline__ float leaky(float x) { return x >= 0.0f ? x : 0.7f * x; }

__device__ __forceinline__ v8f wmma_f16(v16h a, v16h b, v8f c) {
  // (neg_a, A, neg_b, B, c_mod, C, reuse_a, reuse_b)
  return __builtin_amdgcn_wmma_f32_16x16x32_f16(false, a, false, b, (short)0, c,
                                                false, false);
}

__global__ __launch_bounds__(128) void shgnn_target_fused(
    const float* __restrict__ f0, const float* __restrict__ f1,
    const float* __restrict__ f2, const float* __restrict__ W,
    const float* __restrict__ b, const float* __restrict__ Wc,
    const float* __restrict__ bc, const float* __restrict__ Wout,
    const float* __restrict__ bout, const int* __restrict__ deg1,
    const int* __restrict__ deg2, const int* __restrict__ targets,
    float* __restrict__ out_logits, float* __restrict__ out_h) {
  // LDS: 24576 + 16384 + 768 + 768 + 256 + 16384 = 59136 B
  __shared__ _Float16 ldsWt[3 * 64 * 64];   // W transposed: [type][n][k], fp16
  __shared__ _Float16 ldsWoutT[64 * 128];   // Wout transposed: [n][k], fp16
  __shared__ float    ldsB[3 * 64];
  __shared__ float    ldsBc[3 * 64];
  __shared__ float    ldsBout[64];
  __shared__ _Float16 xtile[kWavesPerBlock][16 * 128];  // per-wave h tile

  const int tid = threadIdx.x;

  // ---- cooperative weight staging (transpose + fp16 convert) ----
  for (int idx = tid; idx < 3 * 64 * 64; idx += 128) {
    const int i = idx >> 12, rem = idx & 4095;
    const int n = rem >> 6, k = rem & 63;
    ldsWt[idx] = (_Float16)W[i * 4096 + k * 64 + n];
  }
  for (int idx = tid; idx < 64 * 128; idx += 128) {
    const int n = idx >> 7, k = idx & 127;
    ldsWoutT[idx] = (_Float16)Wout[k * 64 + n];
  }
  if (tid < 192) { ldsB[tid] = b[tid]; ldsBc[tid] = bc[tid]; }
  if (tid < 64)  { ldsBout[tid] = bout[tid]; }
  __syncthreads();

  const int wave = tid >> 5;
  const int lane = tid & 31;
  const int tile = blockIdx.x * kWavesPerBlock + wave;
  if (tile * 16 >= kT) return;  // wave-uniform: EXEC all-ones for WMMA

  const int m  = lane & 15;  // A row / B-C column index this lane covers
  const int hi = lane >> 4;  // lane half-group
  _Float16* xrow = &xtile[wave][0];

  // ---- per-row metadata (row m of this tile; duplicated in both halves) ----
  const int t = targets[tile * 16 + m];
  int ty, p;
  if (t >= 2 * kPerType)  { ty = 2; p = t - 2 * kPerType; }
  else if (t >= kPerType) { ty = 1; p = t - kPerType; }
  else                    { ty = 0; p = t; }
  const int d1 = deg1[t];
  const int d2 = deg2[t];
  const float* fptr = (ty == 0 ? f0 : (ty == 1 ? f1 : f2)) + (long)p * kF;

  // ---- gather this lane's A strip (16x64 tile) and convert to fp16 ----
  // A 16x32 f16 layout: lanes 0-15 hold K={kb..kb+7, 16+kb..16+kb+7}, kb=hi*8
  _Float16 hl[2][16];
#pragma unroll
  for (int kt = 0; kt < 2; ++kt) {
    const int kb = kt * 32 + hi * 8;
    v4f a0 = *(const v4f*)(fptr + kb);
    v4f a1 = *(const v4f*)(fptr + kb + 4);
    v4f a2 = *(const v4f*)(fptr + kb + 16);
    v4f a3 = *(const v4f*)(fptr + kb + 20);
#pragma unroll
    for (int j = 0; j < 4; ++j) {
      hl[kt][j]      = (_Float16)a0[j];
      hl[kt][4 + j]  = (_Float16)a1[j];
      hl[kt][8 + j]  = (_Float16)a2[j];
      hl[kt][12 + j] = (_Float16)a3[j];
    }
  }
  // Type-masked A fragments: rows of other types zeroed, so all three type
  // passes can accumulate into one C (rows are disjoint across types).
  v16h afrag[3][2];
#pragma unroll
  for (int i = 0; i < 3; ++i)
#pragma unroll
    for (int kt = 0; kt < 2; ++kt)
#pragma unroll
      for (int j = 0; j < 16; ++j)
        afrag[i][kt][j] = (ty == i) ? hl[kt][j] : (_Float16)0.0f;

  // ---- stage 1: tf = leaky(f @ W[ty] + b[ty]) via 24 WMMAs ----
#pragma unroll
  for (int nt = 0; nt < 4; ++nt) {
    v8f acc = {};
#pragma unroll
    for (int i = 0; i < 3; ++i) {
      // B 32x16 f16: lane col n=nt*16+m, K = kt*32 + {hi*8+0..7, 16+hi*8+0..7}
      const _Float16* wb = ldsWt + i * 4096 + (nt * 16 + m) * 64 + hi * 8;
#pragma unroll
      for (int kt = 0; kt < 2; ++kt) {
        v8h blo = *(const v8h*)(wb + kt * 32);
        v8h bhi = *(const v8h*)(wb + kt * 32 + 16);
        v16h bfrag = __builtin_shufflevector(blo, bhi, 0, 1, 2, 3, 4, 5, 6, 7, 8, 9,
                                             10, 11, 12, 13, 14, 15);
        acc = wmma_f16(afrag[i][kt], bfrag, acc);
      }
    }
    // C/D: lane covers N=nt*16+m, rows M=r+hi*8; bias needs row M's type
#pragma unroll
    for (int r = 0; r < 8; ++r) {
      const int M = r + hi * 8;
      const int tyM = __shfl(ty, M, 32);
      const float v = leaky(acc[r] + ldsB[tyM * 64 + nt * 16 + m]);
      xrow[M * 128 + nt * 16 + m] = (_Float16)v;
    }
  }

  // ---- degree path: tfc = leaky(Wc[ty][d1] + Wc[ty][100+d2] + bc[ty]) ----
  {
    const float* wc1 = Wc + ((long)ty * (kD1 + kD2) + d1) * 64;
    const float* wc2 = Wc + ((long)ty * (kD1 + kD2) + kD1 + d2) * 64;
#pragma unroll
    for (int s = 0; s < 4; ++s) {
      const int c = hi * 32 + s * 8;
      v4f x0 = *(const v4f*)(wc1 + c);
      v4f x1 = *(const v4f*)(wc1 + c + 4);
      v4f y0 = *(const v4f*)(wc2 + c);
      v4f y1 = *(const v4f*)(wc2 + c + 4);
      v8h hv;
#pragma unroll
      for (int j = 0; j < 4; ++j) {
        hv[j]     = (_Float16)leaky(x0[j] + y0[j] + ldsBc[ty * 64 + c + j]);
        hv[4 + j] = (_Float16)leaky(x1[j] + y1[j] + ldsBc[ty * 64 + c + 4 + j]);
      }
      *(v8h*)(xrow + m * 128 + 64 + c) = hv;
    }
  }

  // ---- stage 2: logits = x(16x128) @ WoutT + bout via 16 WMMAs ----
#pragma unroll
  for (int nt = 0; nt < 4; ++nt) {
    v8f acc = {};
#pragma unroll
    for (int kt = 0; kt < 4; ++kt) {
      const _Float16* ap = xrow + m * 128 + kt * 32 + hi * 8;
      v8h alo = *(const v8h*)ap;          // K = kt*32 + hi*8 + 0..7
      v8h ahi = *(const v8h*)(ap + 16);   // K = kt*32 + 16 + hi*8 + 0..7
      v16h a2 = __builtin_shufflevector(alo, ahi, 0, 1, 2, 3, 4, 5, 6, 7, 8, 9,
                                        10, 11, 12, 13, 14, 15);
      const _Float16* bp = ldsWoutT + (nt * 16 + m) * 128 + kt * 32 + hi * 8;
      v8h blo = *(const v8h*)bp;
      v8h bhi = *(const v8h*)(bp + 16);
      v16h b2 = __builtin_shufflevector(blo, bhi, 0, 1, 2, 3, 4, 5, 6, 7, 8, 9,
                                        10, 11, 12, 13, 14, 15);
      acc = wmma_f16(a2, b2, acc);
    }
#pragma unroll
    for (int r = 0; r < 8; ++r) {
      const int M = r + hi * 8;
      const int N = nt * 16 + m;
      const _Float16 lv = (_Float16)acc[r] + (_Float16)ldsBout[N];  // fp16 math as in ref
      out_logits[(long)(tile * 16 + M) * 64 + N] = (float)lv;
    }
  }

  // ---- h output: 16x128 fp16 tile -> float ----
  {
    float* hb = out_h + (long)(tile * 16 + m) * 128 + hi * 64;
    const _Float16* src = xrow + m * 128 + hi * 64;
#pragma unroll
    for (int s = 0; s < 8; ++s) {
      v8h hv = *(const v8h*)(src + s * 8);
      v4f o0, o1;
#pragma unroll
      for (int j = 0; j < 4; ++j) { o0[j] = (float)hv[j]; o1[j] = (float)hv[4 + j]; }
      *(v4f*)(hb + s * 8)     = o0;
      *(v4f*)(hb + s * 8 + 4) = o1;
    }
  }
}

extern "C" void kernel_launch(void* const* d_in, const int* in_sizes, int n_in,
                              void* d_out, int out_size, void* d_ws, size_t ws_size,
                              hipStream_t stream) {
  const float* f0   = (const float*)d_in[0];
  const float* f1   = (const float*)d_in[1];
  const float* f2   = (const float*)d_in[2];
  const float* W    = (const float*)d_in[3];   // [3][64][64]
  const float* b    = (const float*)d_in[4];   // [3][64]
  const float* Wc   = (const float*)d_in[5];   // [3][200][64]
  const float* bc   = (const float*)d_in[6];   // [3][64]
  const float* Wout = (const float*)d_in[7];   // [128][64]
  const float* bout = (const float*)d_in[8];   // [64]
  // d_in[9] = type_mask: implied by contiguous per-type ranges; unused
  const int* deg1    = (const int*)d_in[10];
  const int* deg2    = (const int*)d_in[11];
  const int* targets = (const int*)d_in[12];

  float* out_logits = (float*)d_out;                    // [T][64]
  float* out_h      = out_logits + (long)kT * kOut;     // [T][128]

  const int tiles  = (kT + 15) / 16;                    // 3125 (exact)
  const int blocks = (tiles + kWavesPerBlock - 1) / kWavesPerBlock;
  shgnn_target_fused<<<blocks, 128, 0, stream>>>(f0, f1, f2, W, b, Wc, bc, Wout,
                                                 bout, deg1, deg2, targets,
                                                 out_logits, out_h);
}
